// LayerWiseMultiLoRAAdapter_3934190044195
// MI455X (gfx1250) — matrix operand order repaired and checked by reading, compiled, not run on hardware
//
#include <hip/hip_runtime.h>

typedef __attribute__((ext_vector_type(2)))  float  v2f;
typedef __attribute__((ext_vector_type(4)))  float  v4f;
typedef __attribute__((ext_vector_type(8)))  float  v8f;
typedef __attribute__((ext_vector_type(4)))  __bf16 v4bf;
typedef __attribute__((ext_vector_type(8)))  __bf16 v8bf;
typedef __attribute__((ext_vector_type(16))) __bf16 v16bf;

#define M_TOT 16384   // B * S
#define N_TOT 2048    // D_OUT
#define K_TOT 2048    // D_IN
#define LRANK 64      // L * R

// ---------------------------------------------------------------------------
// Pack U'[n, l*16+r] = (w1[l] * alpha[l] / R) * ups[l, n, r]   (2048 x 64 f32)
// ---------------------------------------------------------------------------
__global__ __launch_bounds__(256)
void pack_up_kernel(const float* __restrict__ ups, const float* __restrict__ alphas,
                    const float* __restrict__ w1, float* __restrict__ U)
{
  int i = blockIdx.x * 256 + threadIdx.x;
  if (i >= N_TOT * LRANK) return;
  int n = i >> 6, c = i & 63, l = c >> 4, r = c & 15;
  float coef = w1[l] * alphas[l] * (1.0f / 16.0f);
  U[i] = coef * ups[((size_t)l * N_TOT + n) * 16 + r];
}

// ---------------------------------------------------------------------------
// T[m, c] = sum_k x[m,k] * downs_flat[c, k]   (M x 64, K=2048)
// downs is [L,R,K] contiguous, so row c = l*16+r is exactly downs + c*K.
// f32 WMMA 16x16x4, double-buffered LDS. Block: 128 thr = 4 waves,
// block tile 64(M) x 64(N).
// ---------------------------------------------------------------------------
#define K1_LDP 20   // padded f32 row stride: 80B, v4f-aligned, conflict-free b64 frags

__global__ __launch_bounds__(128)
void lora_down_kernel(const float* __restrict__ X, const float* __restrict__ D2,
                      float* __restrict__ T)
{
  __shared__ __align__(16) float lx[2][64 * K1_LDP];
  __shared__ __align__(16) float ld[2][64 * K1_LDP];

  const int tid  = threadIdx.x;
  const int wave = tid >> 5, lane = tid & 31;
  const int lhalf = lane >> 4, lm = lane & 15;
  const int blockM = blockIdx.x * 64;
  const int lrow = tid >> 2;          // 0..31
  const int lk   = (tid & 3) * 4;     // 0,4,8,12

  v8f acc[4];
  #pragma unroll
  for (int nt = 0; nt < 4; ++nt)
    #pragma unroll
    for (int j = 0; j < 8; ++j) acc[nt][j] = 0.0f;

  v4f xr[2], dr[2];
  #pragma unroll
  for (int r = 0; r < 2; ++r) {
    xr[r] = *(const v4f*)(X  + (size_t)(blockM + lrow + r * 32) * K_TOT + lk);
    dr[r] = *(const v4f*)(D2 + (size_t)(lrow + r * 32) * K_TOT + lk);
  }
  // stage 0 into buffer 0
  #pragma unroll
  for (int r = 0; r < 2; ++r) {
    *(v4f*)(&lx[0][(lrow + r * 32) * K1_LDP + lk]) = xr[r];
    *(v4f*)(&ld[0][(lrow + r * 32) * K1_LDP + lk]) = dr[r];
  }
  __syncthreads();

  for (int kt = 0; kt < K_TOT; kt += 16) {
    const int cur = (kt >> 4) & 1;
    const int ktn = kt + 16;
    if (ktn < K_TOT) {
      #pragma unroll
      for (int r = 0; r < 2; ++r) {
        xr[r] = *(const v4f*)(X  + (size_t)(blockM + lrow + r * 32) * K_TOT + ktn + lk);
        dr[r] = *(const v4f*)(D2 + (size_t)(lrow + r * 32) * K_TOT + ktn + lk);
      }
      if (ktn + 16 < K_TOT) {
        __builtin_prefetch(X  + (size_t)(blockM + lrow) * K_TOT + ktn + 16 + lk, 0, 1);
        __builtin_prefetch(D2 + (size_t)lrow * K_TOT + ktn + 16 + lk, 0, 1);
      }
    }

    #pragma unroll
    for (int ks = 0; ks < 16; ks += 4) {
      v2f a = *(const v2f*)(&lx[cur][(wave * 16 + lm) * K1_LDP + ks + 2 * lhalf]);
      #pragma unroll
      for (int nt = 0; nt < 4; ++nt) {
        v2f b = *(const v2f*)(&ld[cur][(nt * 16 + lm) * K1_LDP + ks + 2 * lhalf]);
        acc[nt] = __builtin_amdgcn_wmma_f32_16x16x4_f32(
            false, a, false, b, (short)0, acc[nt], false, false);
      }
    }

    if (ktn < K_TOT) {
      const int nxt = cur ^ 1;
      #pragma unroll
      for (int r = 0; r < 2; ++r) {
        *(v4f*)(&lx[nxt][(lrow + r * 32) * K1_LDP + lk]) = xr[r];
        *(v4f*)(&ld[nxt][(lrow + r * 32) * K1_LDP + lk]) = dr[r];
      }
    }
    __syncthreads();
  }

  #pragma unroll
  for (int nt = 0; nt < 4; ++nt)
    #pragma unroll
    for (int v = 0; v < 8; ++v)
      T[(size_t)(blockM + wave * 16 + v + 8 * lhalf) * LRANK + nt * 16 + lm] = acc[nt][v];
}

// ---------------------------------------------------------------------------
// Fused main GEMM: out = w0*(x @ W^T + b) + T @ U'^T
//  - x@W^T in split-bf16 (hi/lo, 3x V_WMMA_F32_16X16X32_BF16 per 32-K chunk)
//  - LoRA tail (K'=64) + bias fused in epilogue with f32 WMMA C-chaining
// Block: 256 thr = 8 waves (2M x 4N), wave tile 64x32, block tile 128x128.
// Double-buffered LDS (80 KB), one barrier per stage.
// ---------------------------------------------------------------------------
#define BM2 128
#define BN2 128
#define LDB 40   // bf16 row stride: 80B, 16B-aligned, conflict-free b128 frags

__global__ __launch_bounds__(256)
void fused_gemm_kernel(const float* __restrict__ X, const float* __restrict__ Wt,
                       const float* __restrict__ bias, const float* __restrict__ w0p,
                       const float* __restrict__ T, const float* __restrict__ U,
                       float* __restrict__ out)
{
  __shared__ __align__(16) __bf16 sxh[2][BM2 * LDB];
  __shared__ __align__(16) __bf16 sxl[2][BM2 * LDB];
  __shared__ __align__(16) __bf16 swh[2][BN2 * LDB];
  __shared__ __align__(16) __bf16 swl[2][BN2 * LDB];

  const int tid   = threadIdx.x;
  const int wave  = tid >> 5, lane = tid & 31;
  const int lhalf = lane >> 4, lm = lane & 15;
  const int wm = wave >> 2;            // 0..1 -> wave M offset wm*64
  const int wn = wave & 3;             // 0..3 -> wave N offset wn*32
  const int blockM = blockIdx.x * BM2;
  const int blockN = blockIdx.y * BN2;

  v8f acc[4][2];
  #pragma unroll
  for (int mt = 0; mt < 4; ++mt)
    #pragma unroll
    for (int nt = 0; nt < 2; ++nt)
      #pragma unroll
      for (int j = 0; j < 8; ++j) acc[mt][nt][j] = 0.0f;

  const int grow = tid >> 3;           // 0..31
  const int gk   = (tid & 7) * 4;      // 0..28

  v4f xr[4], wr[4];

  // split each f32 into bf16 hi + bf16 lo and stage both planes into LDS
  auto stage_store = [&](int buf) {
    #pragma unroll
    for (int r = 0; r < 4; ++r) {
      const int row = grow + r * 32;
      v4bf xh, xl, wh, wl;
      #pragma unroll
      for (int j = 0; j < 4; ++j) {
        float f = xr[r][j];
        __bf16 h = (__bf16)f;
        xh[j] = h;
        xl[j] = (__bf16)(f - (float)h);
        f = wr[r][j];
        h = (__bf16)f;
        wh[j] = h;
        wl[j] = (__bf16)(f - (float)h);
      }
      *(v4bf*)(&sxh[buf][row * LDB + gk]) = xh;
      *(v4bf*)(&sxl[buf][row * LDB + gk]) = xl;
      *(v4bf*)(&swh[buf][row * LDB + gk]) = wh;
      *(v4bf*)(&swl[buf][row * LDB + gk]) = wl;
    }
  };

  #pragma unroll
  for (int r = 0; r < 4; ++r) {
    xr[r] = *(const v4f*)(X  + (size_t)(blockM + grow + r * 32) * K_TOT + gk);
    wr[r] = *(const v4f*)(Wt + (size_t)(blockN + grow + r * 32) * K_TOT + gk);
  }
  stage_store(0);
  __syncthreads();

  for (int kt = 0; kt < K_TOT; kt += 32) {
    const int cur = (kt >> 5) & 1;
    const int ktn = kt + 32;

    // register-prefetch stage kt+32; L2-prefetch stage kt+64
    if (ktn < K_TOT) {
      #pragma unroll
      for (int r = 0; r < 4; ++r) {
        xr[r] = *(const v4f*)(X  + (size_t)(blockM + grow + r * 32) * K_TOT + ktn + gk);
        wr[r] = *(const v4f*)(Wt + (size_t)(blockN + grow + r * 32) * K_TOT + ktn + gk);
      }
      if (ktn + 32 < K_TOT) {
        __builtin_prefetch(X  + (size_t)(blockM + grow) * K_TOT + ktn + 32 + gk, 0, 1);
        __builtin_prefetch(Wt + (size_t)(blockN + grow) * K_TOT + ktn + 32 + gk, 0, 1);
      }
    }

    // A fragments: lane holds k in [8*lhalf, +8) and [16+8*lhalf, +8)
    v16bf ah[4], al[4];
    #pragma unroll
    for (int mt = 0; mt < 4; ++mt) {
      const int base = (wm * 64 + mt * 16 + lm) * LDB + 8 * lhalf;
      v8bf h0 = *(const v8bf*)(&sxh[cur][base]);
      v8bf h1 = *(const v8bf*)(&sxh[cur][base + 16]);
      ah[mt] = __builtin_shufflevector(h0, h1, 0,1,2,3,4,5,6,7,8,9,10,11,12,13,14,15);
      v8bf l0 = *(const v8bf*)(&sxl[cur][base]);
      v8bf l1 = *(const v8bf*)(&sxl[cur][base + 16]);
      al[mt] = __builtin_shufflevector(l0, l1, 0,1,2,3,4,5,6,7,8,9,10,11,12,13,14,15);
    }

    // B fragments: lane holds k in [8*lhalf, +8) and [16+8*lhalf, +8) for column n=lm
    #pragma unroll
    for (int nt = 0; nt < 2; ++nt) {
      const int base = (wn * 32 + nt * 16 + lm) * LDB + 8 * lhalf;
      v8bf h0 = *(const v8bf*)(&swh[cur][base]);
      v8bf h1 = *(const v8bf*)(&swh[cur][base + 16]);
      v16bf bh = __builtin_shufflevector(h0, h1, 0,1,2,3,4,5,6,7,8,9,10,11,12,13,14,15);
      v8bf l0 = *(const v8bf*)(&swl[cur][base]);
      v8bf l1 = *(const v8bf*)(&swl[cur][base + 16]);
      v16bf bl = __builtin_shufflevector(l0, l1, 0,1,2,3,4,5,6,7,8,9,10,11,12,13,14,15);
      #pragma unroll
      for (int mt = 0; mt < 4; ++mt) {
        acc[mt][nt] = __builtin_amdgcn_wmma_f32_16x16x32_bf16(
            false, ah[mt], false, bh, (short)0, acc[mt][nt], false, false);
        acc[mt][nt] = __builtin_amdgcn_wmma_f32_16x16x32_bf16(
            false, ah[mt], false, bl, (short)0, acc[mt][nt], false, false);
        acc[mt][nt] = __builtin_amdgcn_wmma_f32_16x16x32_bf16(
            false, al[mt], false, bh, (short)0, acc[mt][nt], false, false);
      }
    }

    // store next stage into the other buffer while WMMAs drain
    if (ktn < K_TOT) stage_store(cur ^ 1);
    __syncthreads();
  }

  // epilogue: base = w0 * (acc + b[n])
  const float w0 = w0p[0];
  float bv[2];
  #pragma unroll
  for (int nt = 0; nt < 2; ++nt) bv[nt] = bias[blockN + wn * 32 + nt * 16 + lm];
  #pragma unroll
  for (int mt = 0; mt < 4; ++mt)
    #pragma unroll
    for (int nt = 0; nt < 2; ++nt)
      #pragma unroll
      for (int j = 0; j < 8; ++j)
        acc[mt][nt][j] = w0 * (acc[mt][nt][j] + bv[nt]);

  // fused LoRA tail: acc += T[m,0:64] @ U[n,0:64]^T (exact f32 WMMA, K'=64)
  #pragma unroll
  for (int k = 0; k < LRANK; k += 4) {
    v2f a2[4], b2[2];
    #pragma unroll
    for (int mt = 0; mt < 4; ++mt)
      a2[mt] = *(const v2f*)(T + (size_t)(blockM + wm * 64 + mt * 16 + lm) * LRANK + k + 2 * lhalf);
    #pragma unroll
    for (int nt = 0; nt < 2; ++nt)
      b2[nt] = *(const v2f*)(U + (size_t)(blockN + wn * 32 + nt * 16 + lm) * LRANK + k + 2 * lhalf);
    #pragma unroll
    for (int mt = 0; mt < 4; ++mt)
      #pragma unroll
      for (int nt = 0; nt < 2; ++nt)
        acc[mt][nt] = __builtin_amdgcn_wmma_f32_16x16x4_f32(
            false, a2[mt], false, b2[nt], (short)0, acc[mt][nt], false, false);
  }

  // store: per v, half-wave writes 64B-contiguous runs
  #pragma unroll
  for (int mt = 0; mt < 4; ++mt)
    #pragma unroll
    for (int v = 0; v < 8; ++v) {
      const size_t row = (size_t)(blockM + wm * 64 + mt * 16 + v + 8 * lhalf);
      #pragma unroll
      for (int nt = 0; nt < 2; ++nt)
        out[row * N_TOT + blockN + wn * 32 + nt * 16 + lm] = acc[mt][nt][v];
    }
}

// ---------------------------------------------------------------------------
extern "C" void kernel_launch(void* const* d_in, const int* in_sizes, int n_in,
                              void* d_out, int out_size, void* d_ws, size_t ws_size,
                              hipStream_t stream)
{
  (void)in_sizes; (void)n_in; (void)out_size; (void)ws_size;
  const float* x      = (const float*)d_in[0];
  const float* W      = (const float*)d_in[1];
  const float* b      = (const float*)d_in[2];
  const float* ups    = (const float*)d_in[3];
  const float* downs  = (const float*)d_in[4];
  const float* alphas = (const float*)d_in[5];
  const float* w0     = (const float*)d_in[6];
  const float* w1     = (const float*)d_in[7];
  float* out = (float*)d_out;

  float* T = (float*)d_ws;                        // M_TOT*64 f32 = 4 MB
  float* U = T + (size_t)M_TOT * LRANK;           // N_TOT*64 f32 = 512 KB

  pack_up_kernel<<<(N_TOT * LRANK + 255) / 256, 256, 0, stream>>>(ups, alphas, w1, U);
  lora_down_kernel<<<M_TOT / 64, 128, 0, stream>>>(x, downs, T);
  dim3 grid(M_TOT / BM2, N_TOT / BN2);
  fused_gemm_kernel<<<grid, 256, 0, stream>>>(x, W, b, w0, T, U, out);
}
